// LargeViewFieldSpatialEmbedding_34462817583480
// MI455X (gfx1250) — matrix operand order repaired and checked by reading, not compiled, not run on hardware
//
#include <hip/hip_runtime.h>
#include <hip/hip_bf16.h>
#include <math.h>

// ---------------------------------------------------------------------------
// MI455X / gfx1250 fused LIIF-style decoder, round 2.
//
// Roofline: reference = per-pixel 1x1-conv MLPs (~64 GFLOP fp32), but its
// query-coord bug (q0 = base[...,0:1]+dxs; q1 = q0+dxs) makes the output
// constant along the width -> compute 448 rows with bf16 WMMA (~142 MFLOP),
// then broadcast-store 2.4 MB (HBM-bound, ~0.1us at 23.3 TB/s).
//
// Round-2 changes:
//  * branchless tanh-GELU on v_exp_f32/v_rcp_f32 (kills the divergent erf blob)
//  * A fragments loaded once per layer into registers (VGPR-rich wave32),
//    via plain 16B LDS vector loads, no transpose (2 DS ops vs 16 scalar)
//  * 4 waves/block, N-tiles split across waves; float4 broadcast stores
// ---------------------------------------------------------------------------

typedef __bf16 bf16;
typedef bf16  v16bf __attribute__((ext_vector_type(16)));
typedef float v8f   __attribute__((ext_vector_type(8)));
typedef int   v4i   __attribute__((ext_vector_type(4)));

#define NTAP   9
#define CFEAT  32
#define HF     64
#define WF     64
#define SH     448
#define SW     448
#define EMB    288
#define NWAVES 4

#define FRAG_ELEMS 512          // one 32(K) x 16(N) bf16 fragment
#define L1_OFF 0                // se_w1 : 288x20  -> NT=18, KT=1 (K padded to 32)
#define L2_OFF 18               // se_w2 : 288x288 -> NT=18, KT=9
#define L3_OFF 180              // dec_w1: 128x288 -> NT=8,  KT=9
#define L4_OFF 252              // dec_w2: 128x128 -> NT=8,  KT=4
#define L5_OFF 284              // dec_w3: 128x128 -> NT=8,  KT=4
#define L6_OFF 316              // dec_w4: 3x128   -> NT=1,  KT=4 (N padded to 16)
#define NFRAGS 320
#define FRAG_BYTES (NFRAGS * FRAG_ELEMS * 2)

#define QSTR 17                 // actQ (f32) row stride, odd to stagger banks

// K index inside a 16x32 bf16 fragment for element e (0..15) of a lane.
__device__ __forceinline__ int k_local(int e, int hi) {
    return (e < 8 ? e : e + 8) + (hi ? 8 : 0);
}

// Branchless GELU (tanh form) on hardware transcendentals:
// tanh(y) = 1 - 2/(1+exp2(y*2*log2e))
__device__ __forceinline__ float gelu_fast(float x) {
    float y = 0.7978845608f * (x + 0.044715f * x * x * x);
    float e = __builtin_amdgcn_exp2f(y * 2.8853900818f);   // exp(2y)
    float t = 1.0f - 2.0f * __builtin_amdgcn_rcpf(e + 1.0f);
    return 0.5f * x * (1.0f + t);
}

// ---------------------------------------------------------------------------
// Kernel 1: fp32 weights -> bf16 B-fragments (lane-major, 32B per lane).
// frag[(nt,kt)][lane*16+e] = W[nt*16 + lane%16][kt*32 + k_local(e, lane>=16)]
// ---------------------------------------------------------------------------
__global__ void repack_weights(const float* __restrict__ w1, const float* __restrict__ w2,
                               const float* __restrict__ w3, const float* __restrict__ w4,
                               const float* __restrict__ w5, const float* __restrict__ w6,
                               bf16* __restrict__ frags) {
    int gid = blockIdx.x * blockDim.x + threadIdx.x;
    if (gid >= NFRAGS * 32) return;
    int fragIdx = gid >> 5;
    int lane    = gid & 31;
    int hi      = lane >> 4;
    int n16     = lane & 15;

    const float* W; int Cout, Cin, KT, base;
    if      (fragIdx < L2_OFF) { W = w1; Cout = EMB; Cin = 20;  KT = 1; base = L1_OFF; }
    else if (fragIdx < L3_OFF) { W = w2; Cout = EMB; Cin = EMB; KT = 9; base = L2_OFF; }
    else if (fragIdx < L4_OFF) { W = w3; Cout = 128; Cin = EMB; KT = 9; base = L3_OFF; }
    else if (fragIdx < L5_OFF) { W = w4; Cout = 128; Cin = 128; KT = 4; base = L4_OFF; }
    else if (fragIdx < L6_OFF) { W = w5; Cout = 128; Cin = 128; KT = 4; base = L5_OFF; }
    else                       { W = w6; Cout = 3;   Cin = 128; KT = 4; base = L6_OFF; }

    int rel = fragIdx - base;
    int nt  = rel / KT;
    int kt  = rel % KT;
    int n   = nt * 16 + n16;

    bf16* dst = frags + fragIdx * FRAG_ELEMS + lane * 16;
#pragma unroll
    for (int e = 0; e < 16; ++e) {
        int k = kt * 32 + k_local(e, hi);
        float v = (n < Cout && k < Cin) ? W[n * Cin + k] : 0.0f;
        dst[e] = (bf16)v;
    }
}

// ---------------------------------------------------------------------------
// Activation LDS layout: packed 16x16 bf16 tiles.
// element (row m, channel k) lives at tile=k/16, offset (m*16 + k%16).
// ---------------------------------------------------------------------------
__device__ __forceinline__ void store_act(bf16* out, int ch, int row, float f) {
    out[((ch >> 4) << 8) + row * 16 + (ch & 15)] = (bf16)f;
}

// A fragment (16 rows x 32 K) from two adjacent 16x16 tiles: lane reads row
// m = lane%16, 8 consecutive K at 8*(lane/16) from each tile (16B loads),
// giving k = 8h+s (s<8) and 16+8h+(s-8) (s>=8), matching k_local.
__device__ __forceinline__ v16bf load_a_tr(const bf16* act, int kt, int lane) {
    const bf16* p = act + (2 * kt) * 256 + (lane & 15) * 16 + (lane >> 4) * 8;
    v4i r0 = *(const v4i*)p;
    v4i r1 = *(const v4i*)(p + 256);
    union { v4i q[2]; v16bf v; } u;
    u.q[0] = r0; u.q[1] = r1;
    return u.v;
}

__device__ __forceinline__ v16bf load_b_frag(const bf16* __restrict__ frag, int lane) {
    return *(const v16bf*)(frag + lane * 16);   // 32B contiguous per lane
}

// One MLP layer on a 16-row tile. A fragments are loaded ONCE into registers
// and reused for every N-tile; N-tiles are split across the 4 waves.
// MODE 0: GELU epilogue; MODE 1: elementwise multiply by qmul (f32 LDS).
template<int NT, int KT, int MODE>
__device__ __forceinline__ void mlp_layer(const bf16* __restrict__ in,
                                          const bf16* __restrict__ frags,
                                          const float* __restrict__ bias,
                                          bf16* __restrict__ out,
                                          const float* __restrict__ qmul,
                                          int lane, int wid) {
    int nc = lane & 15;
    int hi = lane >> 4;
    v16bf af[KT];
#pragma unroll
    for (int kt = 0; kt < KT; ++kt) af[kt] = load_a_tr(in, kt, lane);
    for (int nt = wid; nt < NT; nt += NWAVES) {
        float bv = bias[nt * 16 + nc];
        v8f acc;
#pragma unroll
        for (int r = 0; r < 8; ++r) acc[r] = bv;
#pragma unroll
        for (int kt = 0; kt < KT; ++kt) {
            v16bf b = load_b_frag(frags + (nt * KT + kt) * FRAG_ELEMS, lane);
            acc = __builtin_amdgcn_wmma_f32_16x16x32_bf16(false, af[kt], false, b,
                                                          (short)0, acc, false, false);
        }
#pragma unroll
        for (int r = 0; r < 8; ++r) {
            int row = r + hi * 8;           // D row = pixel row in tile
            int ch  = nt * 16 + nc;         // D col = output channel
            float f = acc[r];
            if (MODE == 0) f = gelu_fast(f);
            else           f = f * qmul[ch * QSTR + row];
            store_act(out, ch, row, f);
        }
    }
}

// ---------------------------------------------------------------------------
// Kernel 2: full pipeline for 16 output rows; 4 waves cooperate per block.
// ---------------------------------------------------------------------------
__global__ __launch_bounds__(32 * NWAVES)
void row_pipeline(const float* __restrict__ feat, const bf16* __restrict__ frags,
                  const float* __restrict__ se_b1, const float* __restrict__ se_b2,
                  const float* __restrict__ db1,  const float* __restrict__ db2,
                  const float* __restrict__ db3,  const float* __restrict__ db4,
                  float* __restrict__ rowout) {
    __shared__ alignas(16) bf16  actA[(EMB / 16) * 256];   // 18 tiles, 9216 B
    __shared__ alignas(16) bf16  actB[(EMB / 16) * 256];
    __shared__ alignas(16) float actQ[EMB * QSTR];         // gathered feats, f32

    int tid  = threadIdx.x;
    int lane = tid & 31;
    int wid  = tid >> 5;
    int m    = lane & 15;
    int hi   = lane >> 4;
    int rowBase = blockIdx.x * 16;
    int row  = rowBase + m;

    // LIIF centered coord of this output row (only the row coord matters)
    float ci = (2.0f * (float)row + 1.0f) / 448.0f - 1.0f;

    // --- stage the 20-dim coordinate vector (wave 0, lanes 0-15) ------------
    if (wid == 0 && hi == 0) {
        float xvec[20];
#pragma unroll
        for (int t = 0; t < NTAP; ++t) {
            float dx = (float)(t / 3 - 1);   // dxs = [-1,-1,-1,0,0,0,1,1,1]
            float gy = ci + dx;              // q0 (height)
            float gx = ci + 2.0f * dx;       // q1 (width, post flip)
            int iy = (int)rintf((gy + 1.0f) * 32.0f - 0.5f);
            int ix = (int)rintf((gx + 1.0f) * 32.0f - 0.5f);
            bool valid = (ix >= 0) && (ix < WF) && (iy >= 0) && (iy < HF);
            int iyc = min(max(iy, 0), HF - 1);
            int ixc = min(max(ix, 0), WF - 1);
            float v = valid ? 1.0f : 0.0f;
            float qic0 = v * ((2.0f * (float)iyc + 1.0f) / 64.0f - 1.0f);
            float qic1 = v * ((2.0f * (float)ixc + 1.0f) / 64.0f - 1.0f);
            xvec[2 * t]     = qic0 - gy;
            xvec[2 * t + 1] = qic1 - gx;
        }
        xvec[18] = 1.0f / 64.0f;
        xvec[19] = 1.0f / 64.0f;
#pragma unroll
        for (int k = 0; k < 20; ++k) store_act(actA, k, m, xvec[k]);
#pragma unroll
        for (int k = 20; k < 32; ++k) store_act(actA, k, m, 0.0f);
    }

    // --- gather tap features (taps split across waves) ----------------------
    for (int t = wid; t < NTAP; t += NWAVES) {
        float dx = (float)(t / 3 - 1);
        float gy = ci + dx;
        float gx = ci + 2.0f * dx;
        int iy = (int)rintf((gy + 1.0f) * 32.0f - 0.5f);
        int ix = (int)rintf((gx + 1.0f) * 32.0f - 0.5f);
        bool valid = (ix >= 0) && (ix < WF) && (iy >= 0) && (iy < HF);
        float v = valid ? 1.0f : 0.0f;
        const float* src = feat + min(max(iy, 0), HF - 1) * WF + min(max(ix, 0), WF - 1);
#pragma unroll
        for (int cc = 0; cc < 16; ++cc) {
            int c = hi * 16 + cc;            // lane covers 16 of the 32 channels
            actQ[(t * CFEAT + c) * QSTR + m] = src[c * (HF * WF)] * v;
        }
    }
    __syncthreads();

    // --- spatial-embedding MLP ----------------------------------------------
    mlp_layer<18, 1, 0>(actA, frags + L1_OFF * FRAG_ELEMS, se_b1, actB, actQ, lane, wid);
    __syncthreads();
    mlp_layer<18, 9, 1>(actB, frags + L2_OFF * FRAG_ELEMS, se_b2, actA, actQ, lane, wid);
    __syncthreads();
    // --- decoder MLP --------------------------------------------------------
    mlp_layer<8, 9, 0>(actA, frags + L3_OFF * FRAG_ELEMS, db1, actB, actQ, lane, wid);
    __syncthreads();
    mlp_layer<8, 4, 0>(actB, frags + L4_OFF * FRAG_ELEMS, db2, actA, actQ, lane, wid);
    __syncthreads();
    mlp_layer<8, 4, 0>(actA, frags + L5_OFF * FRAG_ELEMS, db3, actB, actQ, lane, wid);
    __syncthreads();

    // --- final 128 -> 3 (N padded to 16), wave 0 only -----------------------
    if (wid == 0) {
        float bv = (m < 3) ? db4[m] : 0.0f;
        v8f acc;
#pragma unroll
        for (int r = 0; r < 8; ++r) acc[r] = bv;
#pragma unroll
        for (int kt = 0; kt < 4; ++kt) {
            v16bf a = load_a_tr(actB, kt, lane);
            v16bf b = load_b_frag(frags + (L6_OFF + kt) * FRAG_ELEMS, lane);
            acc = __builtin_amdgcn_wmma_f32_16x16x32_bf16(false, a, false, b,
                                                          (short)0, acc, false, false);
        }
        if (m < 3) {
#pragma unroll
            for (int r = 0; r < 8; ++r)
                rowout[m * SH + rowBase + r + hi * 8] = acc[r];
        }
    }
}

// ---------------------------------------------------------------------------
// Kernel 3: broadcast [3][448] row results across width -> [3][448][448].
// float4 stores, fully coalesced: pure HBM bandwidth (2.4 MB).
// ---------------------------------------------------------------------------
__global__ void broadcast_rows(const float* __restrict__ rowout, float4* __restrict__ out) {
    int gid = blockIdx.x * blockDim.x + threadIdx.x;   // exact grid, no bound
    int c   = gid / (SH * (SW / 4));
    int rem = gid - c * (SH * (SW / 4));
    int i   = rem / (SW / 4);
    float v = rowout[c * SH + i];
    out[gid] = make_float4(v, v, v, v);
}

// ---------------------------------------------------------------------------
extern "C" void kernel_launch(void* const* d_in, const int* in_sizes, int n_in,
                              void* d_out, int out_size, void* d_ws, size_t ws_size,
                              hipStream_t stream) {
    const float* feat  = (const float*)d_in[0];
    const float* se_w1 = (const float*)d_in[1];
    const float* se_b1 = (const float*)d_in[2];
    const float* se_w2 = (const float*)d_in[3];
    const float* se_b2 = (const float*)d_in[4];
    const float* dw1   = (const float*)d_in[5];
    const float* db1   = (const float*)d_in[6];
    const float* dw2   = (const float*)d_in[7];
    const float* db2   = (const float*)d_in[8];
    const float* dw3   = (const float*)d_in[9];
    const float* db3   = (const float*)d_in[10];
    const float* dw4   = (const float*)d_in[11];
    const float* db4   = (const float*)d_in[12];
    // d_in[13], d_in[14]: sH = sW = 448 (compile-time constants here)

    bf16*  frags  = (bf16*)d_ws;
    float* rowout = (float*)((char*)d_ws + FRAG_BYTES);

    repack_weights<<<(NFRAGS * 32 + 255) / 256, 256, 0, stream>>>(
        se_w1, se_w2, dw1, dw2, dw3, dw4, frags);
    row_pipeline<<<SH / 16, 32 * NWAVES, 0, stream>>>(
        feat, frags, se_b1, se_b2, db1, db2, db3, db4, rowout);
    broadcast_rows<<<(3 * SH * (SW / 4)) / 256, 256, 0, stream>>>(
        rowout, (float4*)d_out);
}
